// DFWMamba2Block_62208306315805
// MI455X (gfx1250) — matrix operand order repaired and checked by reading, not compiled, not run on hardware
//
#include <hip/hip_runtime.h>

// ---- Problem constants (from reference) ----
#define D_MODEL   1024
#define D_STATE   128
#define D_CONV    4
#define D_INNER   2048
#define NHEADS    32
#define HEADDIM   64
#define D_XBC     2304          // D_INNER + 2*D_STATE
#define D_IN_PROJ 4384          // 2*D_INNER + 2*D_STATE + NHEADS
#define PROJ_LD   4480          // D_IN_PROJ padded to multiple of 128 (35 macro-tiles)
#define BATCH     2
#define SEQ       2048
#define NTOK      (BATCH*SEQ)   // 4096
#define BLK       64            // block_len
#define NCHUNK    (SEQ/BLK)     // 32
#define KKQ       64            // K_SPRAY^2
#define EPSQ      1e-8f

typedef __attribute__((ext_vector_type(16))) __bf16 v16bf;
typedef __attribute__((ext_vector_type(8)))  float  v8f;

// fp32 -> bf16 round-to-nearest-even
__device__ __forceinline__ __bf16 f2bf(float f) {
  unsigned u = __builtin_bit_cast(unsigned, f);
  unsigned r = (u + 0x7fffu + ((u >> 16) & 1u)) >> 16;
  unsigned short s = (unsigned short)r;
  return __builtin_bit_cast(__bf16, s);
}

// 16-bit A-matrix (16x32) k order used here (not AMD's table order): flat element i of v16bf for lane ->
// k = (lane>=16?16:0) + i
__device__ __forceinline__ int kmap_a(int i, int lane) {
  return ((lane >> 4) << 4) | i;
}
// 16-bit B-matrix (32x16), same k order as A here: flat element i -> k = (lane>=16?16:0) + i, col = lane%16

__device__ __forceinline__ float siluf(float v) { return v / (1.f + expf(-v)); }

// gfx1250 async global->LDS copy: each lane moves 16B. Tracked by ASYNCcnt.
__device__ __forceinline__ void async_cp16(unsigned lds_off, const unsigned short* g) {
  asm volatile("global_load_async_to_lds_b128 %0, %1, off"
               :: "v"(lds_off), "v"(g) : "memory");
}
__device__ __forceinline__ unsigned lds_off_of(const void* p) {
  return (unsigned)(unsigned long long)p;   // low 32 bits of generic addr = LDS offset
}

// ---------------- elementwise prep ----------------
__global__ void k_prep_w(const float* __restrict__ lat, const float* __restrict__ scale,
                         unsigned short* __restrict__ out, int rows, int cols) {
  long i = (long)blockIdx.x * blockDim.x + threadIdx.x;
  if (i >= (long)rows * cols) return;
  int r = (int)(i / cols);
  float v = fminf(fmaxf(lat[i], -1.f), 1.f) * scale[r];
  reinterpret_cast<__bf16*>(out)[i] = f2bf(v);
}

__global__ void k_f2bf(const float* __restrict__ in, unsigned short* __restrict__ out, long n) {
  long i = (long)blockIdx.x * blockDim.x + threadIdx.x;
  if (i >= n) return;
  reinterpret_cast<__bf16*>(out)[i] = f2bf(in[i]);
}

// ---------------- LDS-staged bf16 WMMA GEMM: C(MxN) = A(MxK) * W(NxK)^T ----------------
// Block = 256 threads = 8 waves (2x4), macro-tile 64(M) x 128(N), double-buffered
// async global->LDS staging of 32-wide K slices.
__global__ void __launch_bounds__(256) k_gemm_lds(const unsigned short* __restrict__ Aus,
                                                  const unsigned short* __restrict__ Wus,
                                                  float* __restrict__ C,
                                                  int M, int N, int K, int nBlocksN) {
  __shared__ __align__(16) unsigned short sA[2][64 * 32];    // [stage][row][k]
  __shared__ __align__(16) unsigned short sB[2][128 * 32];   // [stage][nrow][k]
  int tid = threadIdx.x;
  int lane = tid & 31;
  int wave = tid >> 5;
  int bm = blockIdx.x / nBlocksN;
  int bn = blockIdx.x % nBlocksN;
  long m0 = (long)bm * 64;
  long n0 = (long)bn * 128;
  int wm = wave >> 2;              // 0..1  -> rows wm*32..
  int wn = wave & 3;               // 0..3  -> cols wn*32..

  // staging assignment: 16B chunks; A: 64 rows x 4 chunks = 256 = 1/thread;
  // B: 128 rows x 4 chunks = 512 = 2/thread.
  int arow = tid >> 2, achk = tid & 3;
  const unsigned short* gA  = Aus + (m0 + arow) * (long)K + achk * 8;
  int br0 = tid >> 2, br1 = (tid + 256) >> 2, bchk = tid & 3;
  const unsigned short* gB0 = Wus + (n0 + br0) * (long)K + bchk * 8;
  const unsigned short* gB1 = Wus + (n0 + br1) * (long)K + bchk * 8;
  unsigned ldsA[2], ldsB0[2], ldsB1[2];
#pragma unroll
  for (int s = 0; s < 2; ++s) {
    ldsA[s]  = lds_off_of(&sA[s][arow * 32 + achk * 8]);
    ldsB0[s] = lds_off_of(&sB[s][br0 * 32 + bchk * 8]);
    ldsB1[s] = lds_off_of(&sB[s][br1 * 32 + bchk * 8]);
  }

  v8f acc[2][2] = {};
  int nk = K >> 5;

  // prefetch stage 0
  async_cp16(ldsA[0],  gA);
  async_cp16(ldsB0[0], gB0);
  async_cp16(ldsB1[0], gB1);

  for (int kk = 0; kk < nk; ++kk) {
    int s = kk & 1;
    if (kk + 1 < nk) {
      int k1 = (kk + 1) << 5;
      async_cp16(ldsA[s ^ 1],  gA  + k1);
      async_cp16(ldsB0[s ^ 1], gB0 + k1);
      async_cp16(ldsB1[s ^ 1], gB1 + k1);
      asm volatile("s_wait_asynccnt 0x3" ::: "memory");   // stage s landed
    } else {
      asm volatile("s_wait_asynccnt 0x0" ::: "memory");
    }
    __syncthreads();

    const __bf16* tA = reinterpret_cast<const __bf16*>(&sA[s][0]);
    const __bf16* tB = reinterpret_cast<const __bf16*>(&sB[s][0]);
    v16bf afr[2], bfr[2];
#pragma unroll
    for (int u = 0; u < 2; ++u) {
      int mr = wm * 32 + u * 16 + (lane & 15);
#pragma unroll
      for (int i = 0; i < 16; ++i) afr[u][i] = tA[mr * 32 + kmap_a(i, lane)];
      int nr = wn * 32 + u * 16 + (lane & 15);
      int kb = (lane >> 4) << 4;
#pragma unroll
      for (int i = 0; i < 16; ++i) bfr[u][i] = tB[nr * 32 + kb + i];
    }
#pragma unroll
    for (int u = 0; u < 2; ++u)
#pragma unroll
      for (int v = 0; v < 2; ++v)
        acc[u][v] = __builtin_amdgcn_wmma_f32_16x16x32_bf16(false, afr[u], false, bfr[v],
                                                            (short)0, acc[u][v], false, false);
    __syncthreads();   // all waves done reading stage s before it is overwritten
  }

  // store 2x2 tiles
#pragma unroll
  for (int u = 0; u < 2; ++u) {
    int mb = (int)m0 + wm * 32 + u * 16 + ((lane >> 4) << 3);
#pragma unroll
    for (int v = 0; v < 2; ++v) {
      int col = (int)n0 + wn * 32 + v * 16 + (lane & 15);
#pragma unroll
      for (int i = 0; i < 8; ++i)
        C[(long)(mb + i) * N + col] = acc[u][v][i];
    }
  }
}

// ---------------- depthwise causal conv (K=4) + SiLU ----------------
__global__ void k_conv(const float* __restrict__ proj, const float* __restrict__ cw,
                       const float* __restrict__ cb, float* __restrict__ xbc) {
  long i = (long)blockIdx.x * blockDim.x + threadIdx.x;
  if (i >= (long)NTOK * D_XBC) return;
  int ch = (int)(i % D_XBC);
  long t = i / D_XBC;
  int l = (int)(t % SEQ);
  float acc = cb[ch];
#pragma unroll
  for (int j = 0; j < D_CONV; ++j) {
    int ls = l - (D_CONV - 1) + j;
    float xv = (ls >= 0) ? proj[(t - (long)(l - ls)) * PROJ_LD + D_INNER + ch] : 0.f;
    acc += cw[ch * D_CONV + j] * xv;
  }
  xbc[i] = siluf(acc);
}

// ---------------- dt = softplus(proj_dt + bias); A_disc = -exp(A_log)*dt ----------------
__global__ void k_dt(const float* __restrict__ proj, const float* __restrict__ dt_bias,
                     const float* __restrict__ A_log, float* __restrict__ dt,
                     float* __restrict__ Adisc) {
  int i = blockIdx.x * blockDim.x + threadIdx.x;
  if (i >= NTOK * NHEADS) return;
  int h = i % NHEADS;
  long t = i / NHEADS;
  float v = proj[t * PROJ_LD + D_INNER + D_XBC + h] + dt_bias[h];
  float sp = (v > 20.f) ? v : log1pf(expf(v));
  dt[i] = sp;
  Adisc[i] = -expf(A_log[h]) * sp;
}

// ---------------- X = x_ssm * dt ----------------
__global__ void k_x(const float* __restrict__ xbc, const float* __restrict__ dt,
                    float* __restrict__ X) {
  long i = (long)blockIdx.x * blockDim.x + threadIdx.x;
  if (i >= (long)NTOK * D_INNER) return;
  int d = (int)(i % D_INNER);
  long t = i / D_INNER;
  X[i] = xbc[t * D_XBC + d] * dt[t * NHEADS + d / HEADDIM];
}

// ---------------- per-(b,chunk,head): cumsum, Y_diag, intra ----------------
__global__ void k_chunk(const float* __restrict__ xbc, const float* __restrict__ X,
                        const float* __restrict__ Adisc, float* __restrict__ Acs_g,
                        float* __restrict__ Y, float* __restrict__ intra) {
  __shared__ float sAcs[BLK];
  __shared__ float sdec[BLK];
  __shared__ float sM[BLK * BLK];
  int blk = blockIdx.x;                 // (b*NCHUNK + c)*NHEADS + h
  int h = blk % NHEADS;
  int c = (blk / NHEADS) % NCHUNK;
  int b = blk / (NHEADS * NCHUNK);
  int lane = threadIdx.x & 31;
  int wave = threadIdx.x >> 5;
  long tok0 = (long)b * SEQ + (long)c * BLK;

  if (threadIdx.x < BLK) sAcs[threadIdx.x] = Adisc[(tok0 + threadIdx.x) * NHEADS + h];
  __syncthreads();
  if (threadIdx.x == 0) {
    float s = 0.f;
    for (int l = 0; l < BLK; ++l) { s += sAcs[l]; sAcs[l] = s; }
  }
  __syncthreads();
  if (threadIdx.x < BLK) {
    sdec[threadIdx.x] = expf(sAcs[BLK - 1] - sAcs[threadIdx.x]);
    Acs_g[(((long)b * NHEADS + h) * NCHUNK + c) * BLK + threadIdx.x] = sAcs[threadIdx.x];
  }
  __syncthreads();

  const float* Crow = xbc + tok0 * D_XBC + D_INNER + D_STATE;  // Cc, row stride D_XBC
  const float* Brow = xbc + tok0 * D_XBC + D_INNER;            // Bc
  const float* Xc   = X + tok0 * D_INNER + (long)h * HEADDIM;  // Xc, row stride D_INNER

  // G = Cc(64x128) @ Bc^T -> masked into sM
  for (int t = wave; t < 16; t += 8) {
    int tm = t >> 2, tn = t & 3;
    v8f acc = {};
    int mrow = tm * 16 + (lane & 15);
    int scol = tn * 16 + (lane & 15);
    for (int k0 = 0; k0 < D_STATE; k0 += 32) {
      v16bf af, bb;
#pragma unroll
      for (int i = 0; i < 16; ++i)
        af[i] = f2bf(Crow[(long)mrow * D_XBC + k0 + kmap_a(i, lane)]);
      int kb = k0 + ((lane >> 4) << 4);
#pragma unroll
      for (int i = 0; i < 16; ++i)
        bb[i] = f2bf(Brow[(long)scol * D_XBC + kb + i]);
      acc = __builtin_amdgcn_wmma_f32_16x16x32_bf16(false, af, false, bb, (short)0, acc, false, false);
    }
    int mb = tm * 16 + ((lane >> 4) << 3);
#pragma unroll
    for (int i = 0; i < 8; ++i) {
      int l = mb + i;
      sM[l * BLK + scol] = (l >= scol) ? acc[i] * expf(sAcs[l] - sAcs[scol]) : 0.f;
    }
  }
  __syncthreads();

  // Y_diag = sM(64x64) @ Xc(64x64)
  for (int t = wave; t < 16; t += 8) {
    int tm = t >> 2, tn = t & 3;
    v8f acc = {};
    int mrow = tm * 16 + (lane & 15);
    int pcol = tn * 16 + (lane & 15);
    for (int k0 = 0; k0 < BLK; k0 += 32) {
      v16bf af, bb;
#pragma unroll
      for (int i = 0; i < 16; ++i)
        af[i] = f2bf(sM[mrow * BLK + k0 + kmap_a(i, lane)]);
      int kb = k0 + ((lane >> 4) << 4);
#pragma unroll
      for (int i = 0; i < 16; ++i)
        bb[i] = f2bf(Xc[(long)(kb + i) * D_INNER + pcol]);
      acc = __builtin_amdgcn_wmma_f32_16x16x32_bf16(false, af, false, bb, (short)0, acc, false, false);
    }
    int mb = tm * 16 + ((lane >> 4) << 3);
#pragma unroll
    for (int i = 0; i < 8; ++i)
      Y[(tok0 + mb + i) * D_INNER + (long)h * HEADDIM + pcol] = acc[i];
  }

  // intra(64p x 128n) = (decay*Xc)^T @ Bc
  long ibase = (long)blk * (HEADDIM * D_STATE);
  for (int t = wave; t < 32; t += 8) {
    int tm = t >> 3, tn = t & 7;
    v8f acc = {};
    int prow = tm * 16 + (lane & 15);
    int ncol = tn * 16 + (lane & 15);
    for (int k0 = 0; k0 < BLK; k0 += 32) {
      v16bf af, bb;
#pragma unroll
      for (int i = 0; i < 16; ++i) {
        int l = k0 + kmap_a(i, lane);
        af[i] = f2bf(Xc[(long)l * D_INNER + prow] * sdec[l]);
      }
      int kb = k0 + ((lane >> 4) << 4);
#pragma unroll
      for (int i = 0; i < 16; ++i)
        bb[i] = f2bf(Brow[(long)(kb + i) * D_XBC + ncol]);
      acc = __builtin_amdgcn_wmma_f32_16x16x32_bf16(false, af, false, bb, (short)0, acc, false, false);
    }
    int pb = tm * 16 + ((lane >> 4) << 3);
#pragma unroll
    for (int i = 0; i < 8; ++i)
      intra[ibase + (long)(pb + i) * D_STATE + ncol] = acc[i];
  }
}

// ---------------- sequential quantized scan over chunks ----------------
__global__ void k_scan(const float* __restrict__ intra, const float* __restrict__ Acs_g,
                       const float* __restrict__ thresholds, float* __restrict__ starts) {
  __shared__ float sR[KKQ];
  __shared__ float red[128];
  int blk = blockIdx.x;                  // (b*NHEADS + h)*HEADDIM + p
  int p = blk % HEADDIM;
  int h = (blk / HEADDIM) % NHEADS;
  int b = blk / (HEADDIM * NHEADS);
  int n = threadIdx.x;                   // 0..127
  float h_run = 0.f, sd_err = 0.f;
  for (int c = 0; c < NCHUNK; ++c) {
    long sbase = (((long)b * NCHUNK + c) * NHEADS + h) * (long)(HEADDIM * D_STATE) + (long)p * D_STATE;
    starts[sbase + n] = h_run;           // pre-update carry
    __syncthreads();
    if (n < KKQ) sR[n] = thresholds[(c & 255) * KKQ + n];
    float cd = expf(Acs_g[(((long)b * NHEADS + h) * NCHUNK + c) * BLK + (BLK - 1)]);
    float hv = cd * h_run + intra[sbase + n];
    red[n] = hv;
    __syncthreads();
    for (int s = 64; s > 0; s >>= 1) { if (n < s) red[n] += red[n + s]; __syncthreads(); }
    float mean = red[0] / 128.f;
    __syncthreads();
    red[n] = hv * hv;
    __syncthreads();
    for (int s = 64; s > 0; s >>= 1) { if (n < s) red[n] += red[n + s]; __syncthreads(); }
    float sumsq = red[0];
    __syncthreads();
    float var = (sumsq - 128.f * mean * mean) / 127.f;
    float sd = fmaxf(sqrtf(fmaxf(var, 0.f)), EPSQ);
    float lo = mean - 3.f * sd;
    float range = fmaxf(6.f * sd, EPSQ);
    float hs = fminf(fmaxf((hv - lo) / range, 0.f), 1.f);
    float hc = fminf(fmaxf(hs + sd_err, 0.f), 1.f);
    int cnt = 0;
#pragma unroll
    for (int j = 0; j < KKQ; ++j) cnt += (sR[j] < hc) ? 1 : 0;   // searchsorted, side='left'
    float hq = (float)cnt / (float)KKQ;
    float e = fminf(fmaxf(hc - hq, -1.f), 1.f);
    sd_err = rintf(e * 128.f) / 128.f;                           // 8-bit RNE error quantization
    h_run = hq * range + lo;
  }
}

// ---------------- Y += Cc @ start^T * exp(Acs) + D_skip * x_ssm ----------------
__global__ void k_yoff(const float* __restrict__ xbc, const float* __restrict__ starts,
                       const float* __restrict__ Acs_g, const float* __restrict__ D_skip,
                       float* __restrict__ Y) {
  __shared__ float sE[BLK];
  int blk = blockIdx.x;
  int h = blk % NHEADS;
  int c = (blk / NHEADS) % NCHUNK;
  int b = blk / (NHEADS * NCHUNK);
  int lane = threadIdx.x & 31;
  int wave = threadIdx.x >> 5;
  long tok0 = (long)b * SEQ + (long)c * BLK;
  if (threadIdx.x < BLK)
    sE[threadIdx.x] = expf(Acs_g[(((long)b * NHEADS + h) * NCHUNK + c) * BLK + threadIdx.x]);
  __syncthreads();
  const float* Crow = xbc + tok0 * D_XBC + D_INNER + D_STATE;
  long sbase = (long)blk * (HEADDIM * D_STATE);
  float dsk = D_skip[h];
  for (int t = wave; t < 16; t += 8) {
    int tm = t >> 2, tn = t & 3;
    v8f acc = {};
    int mrow = tm * 16 + (lane & 15);
    int pcol = tn * 16 + (lane & 15);
    for (int k0 = 0; k0 < D_STATE; k0 += 32) {
      v16bf af, bb;
#pragma unroll
      for (int i = 0; i < 16; ++i)
        af[i] = f2bf(Crow[(long)mrow * D_XBC + k0 + kmap_a(i, lane)]);
      int kb = k0 + ((lane >> 4) << 4);
#pragma unroll
      for (int i = 0; i < 16; ++i)
        bb[i] = f2bf(starts[sbase + (long)pcol * D_STATE + kb + i]);
      acc = __builtin_amdgcn_wmma_f32_16x16x32_bf16(false, af, false, bb, (short)0, acc, false, false);
    }
    int mb = tm * 16 + ((lane >> 4) << 3);
#pragma unroll
    for (int i = 0; i < 8; ++i) {
      int l = mb + i;
      long idx = (tok0 + l) * D_INNER + (long)h * HEADDIM + pcol;
      Y[idx] = Y[idx] + acc[i] * sE[l] + dsk * xbc[(tok0 + l) * D_XBC + (long)h * HEADDIM + pcol];
    }
  }
}

// ---------------- gate (silu(z)) + RMSNorm -> bf16 ----------------
__global__ void k_gate(const float* __restrict__ Y, const float* __restrict__ proj,
                       const float* __restrict__ norm_w, unsigned short* __restrict__ ybus) {
  __shared__ float red[256];
  __bf16* yb = reinterpret_cast<__bf16*>(ybus);
  long t = blockIdx.x;
  float vals[8];
  float ss = 0.f;
#pragma unroll
  for (int j = 0; j < 8; ++j) {
    int d = threadIdx.x + j * 256;
    float z = proj[t * PROJ_LD + d];
    float y = Y[t * D_INNER + d] * siluf(z);
    vals[j] = y; ss += y * y;
  }
  red[threadIdx.x] = ss;
  __syncthreads();
  for (int s = 128; s > 0; s >>= 1) { if (threadIdx.x < s) red[threadIdx.x] += red[threadIdx.x + s]; __syncthreads(); }
  float scale = rsqrtf(red[0] / (float)D_INNER + 1e-5f);
#pragma unroll
  for (int j = 0; j < 8; ++j) {
    int d = threadIdx.x + j * 256;
    yb[t * D_INNER + d] = f2bf(vals[j] * scale * norm_w[d]);
  }
}

// ---------------- host launcher ----------------
extern "C" void kernel_launch(void* const* d_in, const int* in_sizes, int n_in,
                              void* d_out, int out_size, void* d_ws, size_t ws_size,
                              hipStream_t stream) {
  const float* x          = (const float*)d_in[0];
  const float* in_latent  = (const float*)d_in[1];
  const float* in_scale   = (const float*)d_in[2];
  const float* out_latent = (const float*)d_in[3];
  const float* out_scale  = (const float*)d_in[4];
  const float* conv_w     = (const float*)d_in[5];
  const float* conv_b     = (const float*)d_in[6];
  const float* A_log      = (const float*)d_in[7];
  const float* D_skip     = (const float*)d_in[8];
  const float* dt_bias    = (const float*)d_in[9];
  const float* norm_w     = (const float*)d_in[10];
  const float* thresholds = (const float*)d_in[11];
  (void)in_sizes; (void)n_in; (void)out_size; (void)ws_size;

  char* ws = (char*)d_ws;
  size_t off = 0;
  auto take = [&](size_t bytes) -> void* {
    void* p = (void*)(ws + off);
    off += (bytes + 255) & ~(size_t)255;
    return p;
  };
  unsigned short* w_in_b  = (unsigned short*)take((size_t)PROJ_LD * D_MODEL * 2);  // padded rows
  unsigned short* w_out_b = (unsigned short*)take((size_t)D_MODEL * D_INNER * 2);
  unsigned short* x_b     = (unsigned short*)take((size_t)NTOK * D_MODEL * 2);
  float* proj  = (float*)take((size_t)NTOK * PROJ_LD * 4);                          // padded stride
  float* xbc   = (float*)take((size_t)NTOK * D_XBC * 4);
  float* dt    = (float*)take((size_t)NTOK * NHEADS * 4);
  float* Adisc = (float*)take((size_t)NTOK * NHEADS * 4);
  float* X     = (float*)take((size_t)NTOK * D_INNER * 4);
  float* Acs   = (float*)take((size_t)BATCH * NHEADS * NCHUNK * BLK * 4);
  float* intra = (float*)take((size_t)BATCH * NCHUNK * NHEADS * HEADDIM * D_STATE * 4);
  float* starts= (float*)take((size_t)BATCH * NCHUNK * NHEADS * HEADDIM * D_STATE * 4);
  float* Ybuf  = (float*)take((size_t)NTOK * D_INNER * 4);
  unsigned short* yb = (unsigned short*)take((size_t)NTOK * D_INNER * 2);

  long n;
  n = (long)D_IN_PROJ * D_MODEL;
  k_prep_w<<<(int)((n + 255) / 256), 256, 0, stream>>>(in_latent, in_scale, w_in_b, D_IN_PROJ, D_MODEL);
  n = (long)D_MODEL * D_INNER;
  k_prep_w<<<(int)((n + 255) / 256), 256, 0, stream>>>(out_latent, out_scale, w_out_b, D_MODEL, D_INNER);
  n = (long)NTOK * D_MODEL;
  k_f2bf<<<(int)((n + 255) / 256), 256, 0, stream>>>(x, x_b, n);

  {
    int nbn = PROJ_LD / 128;                       // 35
    int blocks = (NTOK / 64) * nbn;                // 64*35
    k_gemm_lds<<<blocks, 256, 0, stream>>>(x_b, w_in_b, proj, NTOK, PROJ_LD, D_MODEL, nbn);
  }

  n = (long)NTOK * D_XBC;
  k_conv<<<(int)((n + 255) / 256), 256, 0, stream>>>(proj, conv_w, conv_b, xbc);
  n = (long)NTOK * NHEADS;
  k_dt<<<(int)((n + 255) / 256), 256, 0, stream>>>(proj, dt_bias, A_log, dt, Adisc);
  n = (long)NTOK * D_INNER;
  k_x<<<(int)((n + 255) / 256), 256, 0, stream>>>(xbc, dt, X);

  k_chunk<<<BATCH * NCHUNK * NHEADS, 256, 0, stream>>>(xbc, X, Adisc, Acs, Ybuf, intra);
  k_scan<<<BATCH * NHEADS * HEADDIM, 128, 0, stream>>>(intra, Acs, thresholds, starts);
  k_yoff<<<BATCH * NCHUNK * NHEADS, 256, 0, stream>>>(xbc, starts, Acs, D_skip, Ybuf);
  k_gate<<<NTOK, 256, 0, stream>>>(Ybuf, proj, norm_w, yb);

  {
    int nbn = D_MODEL / 128;                       // 8
    int blocks = (NTOK / 64) * nbn;                // 512
    k_gemm_lds<<<blocks, 256, 0, stream>>>(yb, w_out_b, (float*)d_out, NTOK, D_MODEL, D_INNER, nbn);
  }
}
